// SimpleRNN_70076686401874
// MI455X (gfx1250) — matrix operand order repaired and checked by reading, compiled, not run on hardware
//
#include <hip/hip_runtime.h>
#include <stdint.h>

#define EMBED   1024
#define HIDDEN  1024
#define BATCH   32
#define SEQ     512
#define M_TOTAL (BATCH * SEQ)   // 16384 rows in the flattened input projection
#define KT      32              // k-slice per WMMA step

typedef __bf16 bf16;
typedef __attribute__((ext_vector_type(16))) bf16  v16bf;
typedef __attribute__((ext_vector_type(8)))  float v8f;
typedef __attribute__((ext_vector_type(4)))  unsigned int u32x4;
typedef __attribute__((ext_vector_type(8)))  unsigned int u32x8;

union FragBF {
    v16bf v;
    struct { u32x4 lo; u32x4 hi; } q;
};

// round-to-nearest-even fp32 -> bf16
__device__ __forceinline__ bf16 f2bf(float f) {
    union { float f; unsigned u; } c; c.f = f;
    unsigned r = c.u + 0x7FFFu + ((c.u >> 16) & 1u);
    union { unsigned short s; bf16 b; } o;
    o.s = (unsigned short)(r >> 16);
    return o.b;
}

// ---- gfx1250 async Global->LDS path (ASYNCcnt) -----------------------------
__device__ __forceinline__ void async_copy_b128(unsigned lds_off, const void* g) {
    asm volatile("global_load_async_to_lds_b128 %0, %1, off"
                 :: "v"(lds_off), "v"((unsigned long long)(size_t)g)
                 : "memory");
}
__device__ __forceinline__ void wait_asynccnt0() {
    asm volatile("s_wait_asynccnt 0x0" ::: "memory");
}
__device__ __forceinline__ void wait_asynccnt2() {
    asm volatile("s_wait_asynccnt 0x2" ::: "memory");
}

// ---------------------------------------------------------------------------
// Kernel 1: convert inputs to bf16, zero h0 and the grid-barrier counter.
// ---------------------------------------------------------------------------
__global__ void prep_kernel(const float* __restrict__ x,
                            const float* __restrict__ Wih,
                            const float* __restrict__ Whh,
                            bf16* __restrict__ xb,
                            bf16* __restrict__ wib,
                            bf16* __restrict__ whb,
                            bf16* __restrict__ h0,
                            unsigned* __restrict__ cnt) {
    size_t i = (size_t)blockIdx.x * blockDim.x + threadIdx.x;
    if (i < (size_t)M_TOTAL * EMBED)  xb[i]  = f2bf(x[i]);
    if (i < (size_t)HIDDEN * EMBED) { wib[i] = f2bf(Wih[i]); whb[i] = f2bf(Whh[i]); }
    if (i < (size_t)BATCH * HIDDEN)   h0[i]  = f2bf(0.0f);
    if (i == 0)                       *cnt   = 0u;
}

// ---------------------------------------------------------------------------
// Kernel 2: input projection  out[m, n] = x[m, :] . W_ih[n, :] + (b_ih+b_hh)[n]
// 64x64 tile per 128-thread (4-wave) block. B tiles async-DMA'd into a
// double-buffered LDS stage; pipeline: issue k+1, s_wait_asynccnt 2, compute k.
// ---------------------------------------------------------------------------
__global__ void __launch_bounds__(128)
xproj_kernel(const bf16* __restrict__ X,      // [M_TOTAL, EMBED] bf16
             const bf16* __restrict__ W,      // [HIDDEN, EMBED]  bf16
             const float* __restrict__ bih,
             const float* __restrict__ bhh,
             float* __restrict__ out) {       // [M_TOTAL, HIDDEN] fp32
    __shared__ __align__(16) bf16 ldsB[2][64 * KT];   // 2 x 4KB B stage

    const int tid  = threadIdx.x;
    const int lane = tid & 31;
    const int wid  = tid >> 5;
    const int row0 = (blockIdx.x << 6) + (wid << 4);
    const int col0 = blockIdx.y << 6;
    const bool hi  = lane >= 16;
    const int  mr  = lane & 15;

    const v8f vzero = {0.f, 0.f, 0.f, 0.f, 0.f, 0.f, 0.f, 0.f};
    v8f acc[4] = {vzero, vzero, vzero, vzero};

    const bf16* Abase = X + (size_t)(row0 + mr) * EMBED;
    const int a0 = hi ? 8 : 0;     // first A chunk K offset (elems)
    const int a1 = hi ? 24 : 16;   // second A chunk K offset

    // async staging: thread t covers rows srow and srow+32, 16B each
    const int srow  = tid >> 2;           // 0..31
    const int skoff = (tid & 3) * 8;      // bf16 elems within k-slice
    const bf16* gB0 = W + (size_t)(col0 + srow) * EMBED + skoff;
    const bf16* gB1 = W + (size_t)(col0 + 32 + srow) * EMBED + skoff;
    const unsigned lbase0 = (unsigned)(size_t)&ldsB[0][0] + (unsigned)tid * 16u;
    const unsigned lbase1 = (unsigned)(size_t)&ldsB[1][0] + (unsigned)tid * 16u;

    // prologue: stage k-tile 0 into buffer 0
    async_copy_b128(lbase0,        gB0);
    async_copy_b128(lbase0 + 2048, gB1);

    for (int i = 0; i < EMBED / KT; ++i) {
        const int k0 = i * KT;
        __syncthreads();   // all waves done reading the buffer we overwrite next
        if (i + 1 < EMBED / KT) {
            const unsigned lb = ((i + 1) & 1) ? lbase1 : lbase0;
            async_copy_b128(lb,        gB0 + k0 + KT);
            async_copy_b128(lb + 2048, gB1 + k0 + KT);
            wait_asynccnt2();          // tile i retired (in-order)
        } else {
            wait_asynccnt0();
        }
        __syncthreads();   // staged tile i visible to all waves

        __builtin_prefetch(Abase + k0 + 256, 0, 0);   // global_prefetch_b8
        FragBF a;
        a.q.lo = *(const u32x4*)(Abase + k0 + a0);
        a.q.hi = *(const u32x4*)(Abase + k0 + a1);

        const bf16* bb = &ldsB[i & 1][0];
#pragma unroll
        for (int j = 0; j < 4; ++j) {
            const bf16* Bp = bb + ((j << 4) + mr) * KT + (hi ? 8 : 0);
            FragBF b;
            b.q.lo = *(const u32x4*)(Bp);
            b.q.hi = *(const u32x4*)(Bp + 16);
            acc[j] = __builtin_amdgcn_wmma_f32_16x16x32_bf16(
                false, a.v, false, b.v, (short)0, acc[j], false, false);
        }
    }

    const int mbase = row0 + (hi ? 8 : 0);
#pragma unroll
    for (int j = 0; j < 4; ++j) {
        const int col = col0 + (j << 4) + mr;
        const float bias = bih[col] + bhh[col];
#pragma unroll
        for (int r = 0; r < 8; ++r)
            out[(size_t)(mbase + r) * HIDDEN + col] = acc[j][r] + bias;
    }
}

// ---------------------------------------------------------------------------
// Kernel 3: persistent recurrent scan.
// 32 blocks x 128 threads; block b owns output columns [32b, 32b+32).
// Its 32x1024 W_hh slice is DMA'd into LDS ONCE by the Tensor Data Mover
// (2D D# descriptor, TENSORcnt), then reused for all 512 steps.
// ---------------------------------------------------------------------------
__global__ void __launch_bounds__(128)
rnn_scan_kernel(const bf16* __restrict__ Whh,   // [HIDDEN, HIDDEN] bf16
                float* __restrict__ out,        // [BATCH, SEQ, HIDDEN] fp32 (xp in place)
                bf16* __restrict__ hbuf,        // 2 x [BATCH, HIDDEN] bf16 (buf0 zeroed)
                unsigned* __restrict__ cnt) {
    __shared__ __align__(16) bf16 wlds[32 * HIDDEN];   // 64 KB W_hh slice

    const int tid  = threadIdx.x;
    const int lane = tid & 31;
    const int wid  = tid >> 5;
    const int nb   = blockIdx.x << 5;      // first global column of this block
    const int mt   = wid & 1;              // batch-row tile: rows 0-15 or 16-31
    const int ntl  = wid >> 1;             // local N tile: 0 or 1
    const bool hi  = lane >= 16;
    const int  mr  = lane & 15;

    // ---- TDM: one 2D tile DMA, Global -> LDS (ISA ch.8 D# layout) ----------
    {
        const unsigned lds_addr = (unsigned)(size_t)&wlds[0];
        const unsigned long long ga =
            (unsigned long long)(size_t)(Whh + (size_t)nb * HIDDEN);
        u32x4 g0;
        g0[0] = 1u;                                   // count=1 (valid user D#)
        g0[1] = lds_addr;                             // lds_addr (bytes)
        g0[2] = (unsigned)(ga & 0xFFFFFFFFull);       // global_addr[31:0]
        g0[3] = (unsigned)((ga >> 32) & 0x01FFFFFFull)// global_addr[56:32]
              | 0x80000000u;                          // type=2 ("image")
        u32x8 g1;
        g1[0] = 0x00010000u;                          // data_size=1 (2B), no mask
        g1[1] = ((unsigned)HIDDEN & 0xFFFFu) << 16;   // tensor_dim0[15:0]=1024
        g1[2] = ((unsigned)HIDDEN >> 16)              // tensor_dim0[31:16]
              | (((unsigned)HIDDEN & 0xFFFFu) << 16); // tensor_dim1[15:0]=1024
        g1[3] = ((unsigned)HIDDEN >> 16)              // tensor_dim1[31:16]
              | ((unsigned)HIDDEN << 16);             // tile_dim0=1024
        g1[4] = 32u;                                  // tile_dim1=32, tile_dim2=0
        g1[5] = (unsigned)HIDDEN;                     // tensor_dim0_stride=1024
        g1[6] = 0u;                                   // stride0 hi | stride1 lo
        g1[7] = (unsigned)((1024ull * 1024ull) >> 16);// tensor_dim1_stride hi
        if (wid == 0) {                               // one wave issues the DMA
            asm volatile("tensor_load_to_lds %0, %1"
                         :: "s"(g0), "s"(g1) : "memory");
            __builtin_amdgcn_s_wait_tensorcnt(0);
        }
    }
    __syncthreads();   // publish LDS tile to all waves

    const int m0    = mt << 4;
    const int ncol  = nb + (ntl << 4) + mr;                      // global column
    const bf16* bls = wlds + (size_t)((ntl << 4) + mr) * HIDDEN + (hi ? 8 : 0);
    const int a0 = hi ? 8 : 0;
    const int a1 = hi ? 24 : 16;
    const int mb = m0 + (hi ? 8 : 0);

    unsigned target = gridDim.x;
    int pp = 0;
    for (int t = 0; t < SEQ; ++t) {
        const bf16* hin  = hbuf + (size_t)pp * (BATCH * HIDDEN);
        bf16*       hout = hbuf + (size_t)(pp ^ 1) * (BATCH * HIDDEN);

        v8f acc = {0.f, 0.f, 0.f, 0.f, 0.f, 0.f, 0.f, 0.f};
        const bf16* ap = hin + (size_t)(m0 + mr) * HIDDEN;
        for (int k0 = 0; k0 < HIDDEN; k0 += KT) {
            FragBF a, b;
            a.q.lo = *(const u32x4*)(ap + k0 + a0);      // global (L2-resident h)
            a.q.hi = *(const u32x4*)(ap + k0 + a1);
            b.q.lo = *(const u32x4*)(bls + k0);          // LDS (ds_load)
            b.q.hi = *(const u32x4*)(bls + k0 + 16);
            acc = __builtin_amdgcn_wmma_f32_16x16x32_bf16(
                false, a.v, false, b.v, (short)0, acc, false, false);
        }

#pragma unroll
        for (int r = 0; r < 8; ++r) {
            const size_t oidx = ((size_t)(mb + r) * SEQ + t) * HIDDEN + ncol;
            const float v = tanhf(acc[r] + out[oidx]);   // xp already holds bias
            out[oidx] = v;
            hout[(size_t)(mb + r) * HIDDEN + ncol] = f2bf(v);
        }

        // device-wide step barrier (all 32 blocks resident -> safe)
        __threadfence();
        __syncthreads();
        if (tid == 0) {
            atomicAdd(cnt, 1u);
            while (__hip_atomic_load(cnt, __ATOMIC_RELAXED, __HIP_MEMORY_SCOPE_AGENT) < target)
                __builtin_amdgcn_s_sleep(1);
        }
        __syncthreads();
        __threadfence();

        target += gridDim.x;
        pp ^= 1;
    }
}

// ---------------------------------------------------------------------------
extern "C" void kernel_launch(void* const* d_in, const int* in_sizes, int n_in,
                              void* d_out, int out_size, void* d_ws, size_t ws_size,
                              hipStream_t stream) {
    const float* x   = (const float*)d_in[0];
    const float* Wih = (const float*)d_in[1];
    const float* Whh = (const float*)d_in[2];
    const float* bih = (const float*)d_in[3];
    const float* bhh = (const float*)d_in[4];
    float* out = (float*)d_out;

    // workspace layout (bytes), ~37.9 MB total
    char* ws = (char*)d_ws;
    bf16*     xb   = (bf16*)(ws);                         // 32 MB  x in bf16
    bf16*     wib  = (bf16*)(ws + 33554432);              //  2 MB  W_ih bf16
    bf16*     whb  = (bf16*)(ws + 35651584);              //  2 MB  W_hh bf16
    bf16*     hbuf = (bf16*)(ws + 37748736);              // 128 KB h double buffer
    unsigned* cnt  = (unsigned*)(ws + 37748736 + 131072); //  4 B   barrier counter

    prep_kernel<<<(M_TOTAL * EMBED) / 256, 256, 0, stream>>>(
        x, Wih, Whh, xb, wib, whb, hbuf, cnt);

    dim3 g1(M_TOTAL / 64, HIDDEN / 64);
    xproj_kernel<<<g1, 128, 0, stream>>>(xb, wib, bih, bhh, out);

    rnn_scan_kernel<<<32, 128, 0, stream>>>(whb, out, hbuf, cnt);
}
